// MaskUnitAttention_3298534883985
// MI455X (gfx1250) — matrix operand order repaired and checked by reading, not compiled, not run on hardware
//
#include <hip/hip_runtime.h>
#include <hip/hip_bf16.h>

// ---------------------------------------------------------------------------
// MaskUnitAttention for MI455X (gfx1250), wave32 + bf16 WMMA (f32 accumulate)
// + double-buffered Tensor Data Mover (tensor_load_to_lds) LDS staging.
//
// Kernel 1: per (b,mu,head) workgroup fuses QKV projection (gathered head
//           columns of w_qkv), q max-pool (1,2,2), softmax(q k^T / 8) v,
//           writing o[bm,72,768] as bf16 into workspace.
// Kernel 2: o @ w_proj + b_proj as a tiled bf16-WMMA GEMM -> f32 output.
// ---------------------------------------------------------------------------

typedef __attribute__((ext_vector_type(16))) __bf16 v16bf;
typedef __attribute__((ext_vector_type(8)))  float  v8f;

#define DIMX      384
#define W_STRIDE  2304       // 3*DIM_OUT
#define DIM_OUT   768
#define NHEADS    12
#define HD        64
#define T_TOK     288
#define TQ_TOK    72
#define TQ_PAD    80
#define BM_COUNT  256        // B*MU

// LDS layout for kernel 1 (byte offsets into dynamic smem)
#define OFF_KBF   0                         // k  [288][64] bf16   (36864 B)
#define OFF_VBFT  (OFF_KBF  + 288*64*2)     // v^T [64][288] bf16  (36864 B)
#define OFF_QBF   (OFF_VBFT + 64*288*2)     // q  [80][64] bf16    (10240 B)
#define OFF_SCR   (OFF_QBF  + 80*64*2)      // scratch region
// scratch phase A: xbuf0/xbuf1 [288][32] f32 (2x36864) + x_bf [288][32] bf16
//                  (18432) + wT [64][32] bf16 (4096) = 96256
//          comp-0 writeback: q_scratch [288][64] f32 (73728, aliases xbufs)
//          phase B: S [80][288] f32 (92160) then P [80][288] bf16 (+46080)
#define OFF_XB0   (OFF_SCR)
#define OFF_XB1   (OFF_SCR + 288*32*4)
#define OFF_XBF   (OFF_SCR + 2*288*32*4)
#define OFF_WCH   (OFF_XBF + 288*32*2)
#define SMEM1     (OFF_SCR + 80*288*4 + 80*288*2)   // 222208 bytes < 320 KB

#define ASZ2      (128*32*2)                // kernel-2 A tile bytes
#define SMEM2     (2*ASZ2 + 64*32*2)        // 2x As + BsT = 20480 B

// ---- Tensor Data Mover staging (guarded; falls back to manual copies) -----

#if __has_builtin(__builtin_amdgcn_tensor_load_to_lds) && \
    __has_builtin(__builtin_amdgcn_s_wait_tensorcnt)
#define USE_TDM 1
#else
#define USE_TDM 0
#endif

#if USE_TDM
typedef __attribute__((ext_vector_type(4))) unsigned int u32x4;
typedef __attribute__((ext_vector_type(8))) int          i32x8;
typedef __attribute__((ext_vector_type(4))) int          i32x4;

// 2-D tile DMA: global (row-major, stride_elems between rows) -> LDS packed.
// ds_code: 0=1B, 1=2B, 2=4B elements. Issue once per workgroup (EXEC ignored).
__device__ __forceinline__ void tdm_load_2d(const void* gaddr, unsigned lds_byte,
                                            unsigned ds_code,
                                            unsigned tile_x, unsigned tile_y,
                                            unsigned tensor_x, unsigned tensor_y,
                                            unsigned stride_elems) {
  const unsigned long long ga = (unsigned long long)(uintptr_t)gaddr;
  u32x4 g0;
  g0[0] = 1u;                                   // count=1, user-mode D#
  g0[1] = lds_byte;                             // lds_addr (bytes)
  g0[2] = (unsigned)ga;                         // global_addr[31:0]
  g0[3] = (unsigned)((ga >> 32) & 0x01FFFFFFu)  // global_addr[56:32]
        | (2u << 30);                           // type=2 ("image")
  i32x8 g1;
  g1[0] = (int)(ds_code << 16);                 // wg_mask=0, data_size
  g1[1] = (int)((tensor_x & 0xFFFFu) << 16);    // tensor_dim0[15:0] @48
  g1[2] = (int)((tensor_x >> 16) | ((tensor_y & 0xFFFFu) << 16));
  g1[3] = (int)((tensor_y >> 16) | (tile_x << 16));   // tile_dim0 @112
  g1[4] = (int)(tile_y & 0xFFFFu);              // tile_dim1; tile_dim2=0 (2D)
  g1[5] = (int)stride_elems;                    // tensor_dim0_stride[31:0]
  g1[6] = 0;
  g1[7] = 0;
  i32x4 z4; z4[0] = z4[1] = z4[2] = z4[3] = 0;  // groups 2/3 unused (2D)
  i32x8 z8;
#pragma unroll
  for (int i = 0; i < 8; ++i) z8[i] = 0;        // second descriptor: unused
  // clang-23 / therock-10.0 signature: 6 args (extra int32x8 before cpol)
  __builtin_amdgcn_tensor_load_to_lds(g0, g1, z4, z4, z8, 0);
}
#endif

// ---- WMMA helpers ---------------------------------------------------------

__device__ __forceinline__ v8f wmma_bf16(v16bf a, v16bf b, v8f c) {
  // emits v_wmma_f32_16x16x32_bf16
  return __builtin_amdgcn_wmma_f32_16x16x32_bf16(
      /*neg_a=*/false, a, /*neg_b=*/false, b,
      /*c_mod=*/(short)0, c, /*reuse_a=*/false, /*reuse_b=*/false);
}

__device__ __forceinline__ void ld_pair(const __bf16* p, __bf16& lo, __bf16& hi) {
  union { unsigned u; __bf16 h[2]; } cv;
  cv.u = *(const unsigned*)p;   // p is 4B aligned (even element offset)
  lo = cv.h[0]; hi = cv.h[1];
}

// A fragment: 16x32 bf16 tile, row-major (K contiguous), row stride in elems.
// Lane layout: m = lane&15; half = lane>>4; VGPR i pairs:
//   k0 = (i>>2)*16 + half*8 + (i&3)*2
__device__ __forceinline__ v16bf load_a_frag(const __bf16* base, int rs, int lane) {
  const int m = lane & 15, half = lane >> 4;
  const __bf16* rp = base + m * rs;
  v16bf f;
#pragma unroll
  for (int i = 0; i < 8; ++i) {
    const int k0 = ((i >> 2) << 4) + (half << 3) + ((i & 3) << 1);
    __bf16 a, b; ld_pair(rp + k0, a, b);
    f[2 * i] = a; f[2 * i + 1] = b;
  }
  return f;
}

// B fragment: 32x16 bf16 (KxN), supplied in TRANSPOSED storage [n][k]
// (K contiguous). Lane layout: n = lane&15; half = lane>>4; VGPR i:
//   k0 = (i>>2)*16 + half*8 + (i&3)*2
__device__ __forceinline__ v16bf load_b_fragT(const __bf16* baseT, int rs, int lane) {
  const int n = lane & 15, half = lane >> 4;
  const __bf16* rp = baseT + n * rs;
  v16bf f;
#pragma unroll
  for (int i = 0; i < 8; ++i) {
    const int k0 = ((i >> 2) << 4) + (half << 3) + ((i & 3) << 1);
    __bf16 a, b; ld_pair(rp + k0, a, b);
    f[2 * i] = a; f[2 * i + 1] = b;
  }
  return f;
}

// ---- Kernel 1: fused QKV + pool + attention per (b,mu,head) ---------------

__global__ void mua_qkv_attn_kernel(const float* __restrict__ x,
                                    const float* __restrict__ w_qkv,
                                    __bf16* __restrict__ o_buf) {
  extern __shared__ char smem[];
  __bf16* k_bf   = (__bf16*)(smem + OFF_KBF);    // [288][64]
  __bf16* v_bfT  = (__bf16*)(smem + OFF_VBFT);   // [64][288]
  __bf16* q_bf   = (__bf16*)(smem + OFF_QBF);    // [80][64]
  __bf16* x_bf   = (__bf16*)(smem + OFF_XBF);    // [288][32] bf16
  __bf16* w_chT  = (__bf16*)(smem + OFF_WCH);    // [64][32]
  float*  q_scr  = (float*) (smem + OFF_SCR);    // [288][64] (aliases xbufs)
  float*  S_f    = (float*) (smem + OFF_SCR);    // [80][288]
  __bf16* P_bf   = (__bf16*)(smem + OFF_SCR + 80 * 288 * 4);  // [80][288]

  const int tid  = threadIdx.x;
  const int lane = tid & 31;
  const int wave = tid >> 5;
  const int bm   = blockIdx.x / NHEADS;
  const int h    = blockIdx.x % NHEADS;

  const float* xg = x + (size_t)bm * T_TOK * DIMX;

  // ---------------- Phase A: qkv_head = x(288x384) @ w_head(384x64) x3 -----
  for (int comp = 0; comp < 3; ++comp) {
    v8f acc[9];
#pragma unroll
    for (int j = 0; j < 9; ++j)
#pragma unroll
      for (int e = 0; e < 8; ++e) acc[j][e] = 0.0f;

#if USE_TDM
    // prologue: DMA chunk 0 into xbuf0 (TENSORcnt-tracked)
    if (wave == 0)
      tdm_load_2d(xg, __builtin_amdgcn_groupstaticsize() + OFF_XB0,
                  2u, 32u, T_TOK, DIMX, T_TOK, DIMX);
#endif
    for (int kk = 0; kk < DIMX / 32; ++kk) {      // 12 K-steps of 32
#if USE_TDM
      const int cur = kk & 1;
      if (wave == 0) __builtin_amdgcn_s_wait_tensorcnt(0);  // xbuf[cur] ready
      __syncthreads();   // publishes DMA; x_bf free (prev compute done)
      // prefetch next chunk into the other buffer (flies during compute)
      if (wave == 0 && kk + 1 < DIMX / 32)
        tdm_load_2d(xg + (kk + 1) * 32,
                    __builtin_amdgcn_groupstaticsize() +
                        (cur ? OFF_XB0 : OFF_XB1),
                    2u, 32u, T_TOK, DIMX, T_TOK, DIMX);
      // one-shot cooperative convert: xbuf[cur] f32 -> x_bf bf16 (packed)
      {
        const float2* src =
            (const float2*)(smem + (cur ? OFF_XB1 : OFF_XB0));
        unsigned* dst = (unsigned*)x_bf;
        for (int i = tid; i < T_TOK * 32 / 2; i += 256) {
          const float2 f = src[i];
          union { unsigned u; __bf16 hh[2]; } cv;
          cv.hh[0] = (__bf16)f.x; cv.hh[1] = (__bf16)f.y;
          dst[i] = cv.u;
        }
      }
#else
      __syncthreads();
      for (int i = tid; i < T_TOK * 32; i += 256) {
        const int t = i >> 5, c = i & 31;
        x_bf[i] = (__bf16)xg[(size_t)t * DIMX + kk * 32 + c];
      }
#endif
      // stage w chunk transposed [d][k]; column = d*36 + h*3 + comp
      for (int i = tid; i < HD * 32; i += 256) {
        const int d = i >> 5, k = i & 31;
        w_chT[i] = (__bf16)w_qkv[(size_t)(kk * 32 + k) * W_STRIDE +
                                 d * 36 + h * 3 + comp];
      }
      __syncthreads();
      // 18 x 4 = 72 C-tiles of 16x16; wave owns idx = wave + 8*j
      // (ni = wave&3 constant per wave -> B fragment hoisted by scheduler)
#pragma unroll
      for (int j = 0; j < 9; ++j) {
        const int idx = wave + 8 * j;
        const int mi = idx >> 2, ni = idx & 3;
        v16bf a = load_a_frag(x_bf + mi * 16 * 32, 32, lane);
        v16bf b = load_b_fragT(w_chT + ni * 16 * 32, 32, lane);
        acc[j] = wmma_bf16(a, b, acc[j]);
      }
    }
    __syncthreads();

    // write back C(288x64): comp 0 -> q_scr f32, 1 -> k_bf, 2 -> v^T
#pragma unroll
    for (int j = 0; j < 9; ++j) {
      const int idx = wave + 8 * j;
      const int mi = idx >> 2, ni = idx & 3;
      const int m0 = mi * 16 + ((lane >> 4) << 3);
      const int n  = ni * 16 + (lane & 15);
#pragma unroll
      for (int r = 0; r < 8; ++r) {
        const int m = m0 + r;
        const float v = acc[j][r];
        if (comp == 0)      q_scr[m * HD + n]    = v;
        else if (comp == 1) k_bf [m * HD + n]    = (__bf16)v;
        else                v_bfT[n * T_TOK + m] = (__bf16)v;
      }
    }
    __syncthreads();

    if (comp == 0) {
      // max-pool q over (1,2,2): t = z*144 + y*24 + dy*12 + x*2 + dx
      for (int i = tid; i < TQ_PAD * HD; i += 256) {
        const int tq = i >> 6, d = i & 63;
        if (tq < TQ_TOK) {
          const int z = tq / 36, rem = tq % 36, y = rem / 6, xx = rem % 6;
          const int t0 = z * 144 + y * 24 + xx * 2;
          float m = q_scr[(t0     ) * HD + d];
          m = fmaxf(m, q_scr[(t0 + 1 ) * HD + d]);
          m = fmaxf(m, q_scr[(t0 + 12) * HD + d]);
          m = fmaxf(m, q_scr[(t0 + 13) * HD + d]);
          q_bf[i] = (__bf16)m;
        } else {
          q_bf[i] = (__bf16)0.0f;       // pad rows 72..79
        }
      }
      __syncthreads();
    }
  }

  // ---------------- Phase B: S = (q k^T) / 8, softmax ----------------------
  // S tiles: 5 (M=80) x 18 (N=288), K = 64 (2 steps)
  for (int t = wave; t < 5 * 18; t += 8) {
    const int mi = t / 18, ni = t % 18;
    v8f acc;
#pragma unroll
    for (int e = 0; e < 8; ++e) acc[e] = 0.0f;
#pragma unroll
    for (int kc = 0; kc < 2; ++kc) {
      v16bf a = load_a_frag(q_bf + mi * 16 * HD + kc * 32, HD, lane);
      v16bf b = load_b_fragT(k_bf + ni * 16 * HD + kc * 32, HD, lane);
      acc = wmma_bf16(a, b, acc);
    }
    const int m0 = mi * 16 + ((lane >> 4) << 3);
    const int n  = ni * 16 + (lane & 15);
#pragma unroll
    for (int r = 0; r < 8; ++r)
      S_f[(m0 + r) * T_TOK + n] = acc[r] * 0.125f;
  }
  __syncthreads();

  // row softmax (one row per thread; rows 72..79 -> zero probabilities)
  if (tid < TQ_PAD) {
    const int r = tid;
    if (r < TQ_TOK) {
      float mx = -1e30f;
      for (int c = 0; c < T_TOK; ++c) mx = fmaxf(mx, S_f[r * T_TOK + c]);
      float sum = 0.0f;
      for (int c = 0; c < T_TOK; ++c) {
        const float e = __expf(S_f[r * T_TOK + c] - mx);
        sum += e;
        S_f[r * T_TOK + c] = e;
      }
      const float inv = 1.0f / sum;
      for (int c = 0; c < T_TOK; ++c)
        P_bf[r * T_TOK + c] = (__bf16)(S_f[r * T_TOK + c] * inv);
    } else {
      for (int c = 0; c < T_TOK; ++c) P_bf[r * T_TOK + c] = (__bf16)0.0f;
    }
  }
  __syncthreads();

  // ---------------- Phase C: O = P(80x288) @ V(288x64) ---------------------
  // O tiles: 5 x 4; K = 288 (9 steps). v supplied transposed ([d][t]).
  for (int t = wave; t < 5 * 4; t += 8) {
    const int mi = t >> 2, ni = t & 3;
    v8f acc;
#pragma unroll
    for (int e = 0; e < 8; ++e) acc[e] = 0.0f;
#pragma unroll
    for (int kc = 0; kc < 9; ++kc) {
      v16bf a = load_a_frag(P_bf + mi * 16 * T_TOK + kc * 32, T_TOK, lane);
      v16bf b = load_b_fragT(v_bfT + ni * 16 * T_TOK + kc * 32, T_TOK, lane);
      acc = wmma_bf16(a, b, acc);
    }
    const int m0 = mi * 16 + ((lane >> 4) << 3);
    const int n  = ni * 16 + (lane & 15);
#pragma unroll
    for (int r = 0; r < 8; ++r) {
      const int m = m0 + r;
      if (m < TQ_TOK)
        o_buf[((size_t)bm * TQ_TOK + m) * DIM_OUT + h * HD + n] =
            (__bf16)acc[r];
    }
  }
}

// ---- Kernel 2: out = o_buf(18432x768) @ w_proj(768x768) + b_proj ----------

__global__ void mua_proj_kernel(const __bf16* __restrict__ o_buf,
                                const float* __restrict__ w_proj,
                                const float* __restrict__ b_proj,
                                float* __restrict__ out) {
  extern __shared__ char smem2[];
  __bf16* As  = (__bf16*)(smem2);                  // 2x [128][32] (row-major)
  __bf16* BsT = (__bf16*)(smem2 + 2 * ASZ2);       // [64][32] (transposed)

  const int tid  = threadIdx.x;
  const int lane = tid & 31;
  const int wave = tid >> 5;
  const int rowbase = blockIdx.x * 128;
  const int colbase = blockIdx.y * 64;

  v8f acc[4];
#pragma unroll
  for (int j = 0; j < 4; ++j)
#pragma unroll
    for (int e = 0; e < 8; ++e) acc[j][e] = 0.0f;

#if USE_TDM
  // prologue: DMA A tile for kc=0 into As[0] (bf16, no conversion needed)
  if (wave == 0)
    tdm_load_2d(o_buf + (size_t)rowbase * DIM_OUT,
                __builtin_amdgcn_groupstaticsize() + 0,
                1u, 32u, 128u, DIM_OUT, BM_COUNT * TQ_TOK, DIM_OUT);
#endif
  for (int kc = 0; kc < DIM_OUT / 32; ++kc) {    // 24 K-steps
#if USE_TDM
    const int cur = kc & 1;
    if (wave == 0) __builtin_amdgcn_s_wait_tensorcnt(0);  // As[cur] ready
    __syncthreads();   // prev compute done -> As[1-cur], BsT free
    if (wave == 0 && kc + 1 < DIM_OUT / 32)
      tdm_load_2d(o_buf + (size_t)rowbase * DIM_OUT + (kc + 1) * 32,
                  __builtin_amdgcn_groupstaticsize() + (cur ? 0 : ASZ2),
                  1u, 32u, 128u, DIM_OUT, BM_COUNT * TQ_TOK, DIM_OUT);
    const __bf16* Acur = As + cur * (128 * 32);
#else
    __syncthreads();
    for (int i = tid; i < 128 * 32; i += 256) {
      const int r = i >> 5, k = i & 31;
      As[i] = o_buf[(size_t)(rowbase + r) * DIM_OUT + kc * 32 + k];
    }
    const __bf16* Acur = As;
#endif
    for (int i = tid; i < 64 * 32; i += 256) {
      const int n = i >> 5, k = i & 31;
      BsT[i] = (__bf16)w_proj[(size_t)(kc * 32 + k) * DIM_OUT + colbase + n];
    }
    __syncthreads();
    // 8 x 4 = 32 C-tiles; wave owns idx = wave + 8*j
#pragma unroll
    for (int j = 0; j < 4; ++j) {
      const int idx = wave + 8 * j;
      const int mi = idx >> 2, ni = idx & 3;
      v16bf a = load_a_frag(Acur + mi * 16 * 32, 32, lane);
      v16bf b = load_b_fragT(BsT + ni * 16 * 32, 32, lane);
      acc[j] = wmma_bf16(a, b, acc[j]);
    }
  }

#pragma unroll
  for (int j = 0; j < 4; ++j) {
    const int idx = wave + 8 * j;
    const int mi = idx >> 2, ni = idx & 3;
    const int m0 = rowbase + mi * 16 + ((lane >> 4) << 3);
    const int n  = colbase + ni * 16 + (lane & 15);
    const float bias = b_proj[n];
#pragma unroll
    for (int r = 0; r < 8; ++r)
      out[(size_t)(m0 + r) * DIM_OUT + n] = acc[j][r] + bias;
  }
}

// ---- Host launcher --------------------------------------------------------

extern "C" void kernel_launch(void* const* d_in, const int* in_sizes, int n_in,
                              void* d_out, int out_size, void* d_ws, size_t ws_size,
                              hipStream_t stream) {
  const float* x      = (const float*)d_in[0];   // [2,128,288,384]
  const float* w_qkv  = (const float*)d_in[1];   // [384,2304]
  const float* w_proj = (const float*)d_in[2];   // [768,768]
  const float* b_proj = (const float*)d_in[3];   // [768]
  float* out = (float*)d_out;                    // [2,128,72,768]

  // workspace: o (pre-proj) as bf16, [bm=256][72][768] = 28.3 MB
  __bf16* o_buf = (__bf16*)d_ws;

  (void)in_sizes; (void)n_in; (void)out_size; (void)ws_size;

  (void)hipFuncSetAttribute((const void*)mua_qkv_attn_kernel,
                            hipFuncAttributeMaxDynamicSharedMemorySize, SMEM1);
  (void)hipFuncSetAttribute((const void*)mua_proj_kernel,
                            hipFuncAttributeMaxDynamicSharedMemorySize, SMEM2);

  // Kernel 1: one workgroup per (b, mu, head) = 3072 blocks
  mua_qkv_attn_kernel<<<BM_COUNT * NHEADS, 256, SMEM1, stream>>>(x, w_qkv, o_buf);

  // Kernel 2: 128x64 output tiles over [18432 x 768]
  dim3 grid2((BM_COUNT * TQ_TOK) / 128, DIM_OUT / 64);
  mua_proj_kernel<<<grid2, 256, SMEM2, stream>>>(o_buf, w_proj, b_proj, out);
}
